// Block_11716670783982
// MI455X (gfx1250) — matrix operand order repaired and checked by reading, not compiled, not run on hardware
//
#include <hip/hip_runtime.h>
#include <hip/hip_bf16.h>
#include <math.h>

// RWKV-v6 block for MI455X (gfx1250, wave32).
// All large matmuls go through one bf16 WMMA GEMM (v_wmma_f32_16x16x32_bf16)
// with fused epilogues; WKV scan is a per-(b,h) sequential VALU kernel.

constexpr int kB  = 4;
constexpr int kT  = 2048;
constexpr int kC  = 2048;
constexpr int kH  = 32;
constexpr int kHS = 64;
constexpr int kFF = 7168;
constexpr int kM  = kB * kT;                   // 8192 rows
constexpr size_t kMC = (size_t)kM * kC;        // one [M,C] fp32 buffer

typedef __attribute__((ext_vector_type(16))) __bf16 v16bf;
typedef __attribute__((ext_vector_type(8)))  float  v8f;

#define EPI_NONE    0
#define EPI_TANH    1
#define EPI_SILU    2
#define EPI_SIGMOID 3
#define EPI_RELUSQ  4
#define EPI_EEW     5   // exp(-exp(td[n] + x))
#define EPI_RESADD  6   // e0[m,n] + x
#define EPI_FINAL   7   // e0[m,n] + e1[m,n] * x

__device__ __forceinline__ unsigned short f2bf(float f) {
  return (unsigned short)(__builtin_bit_cast(unsigned int, f) >> 16);  // truncate
}
__device__ __forceinline__ __bf16 bf_bits(unsigned short s) {
  return __builtin_bit_cast(__bf16, s);
}
// Pack two fp32 -> two bf16 in one dword via v_perm_b32 (high halves).
__device__ __forceinline__ unsigned int pack_bf16(float lo, float hi) {
  return __builtin_amdgcn_perm(__builtin_bit_cast(unsigned int, hi),
                               __builtin_bit_cast(unsigned int, lo),
                               0x07060302u);
}
// Pack float4 -> 4 bf16, store 8 bytes to LDS (address must be 8B aligned).
__device__ __forceinline__ void st4bf(unsigned short* p, float4 v) {
  *(uint2*)(void*)p = make_uint2(pack_bf16(v.x, v.y), pack_bf16(v.z, v.w));
}

template <int EPI>
__device__ __forceinline__ float apply_epi(float x, int gm, int gn, int N,
                                           const float* __restrict__ e0,
                                           const float* __restrict__ e1) {
  if constexpr (EPI == EPI_TANH)    return tanhf(x);
  if constexpr (EPI == EPI_SILU)    return x / (1.f + __expf(-x));
  if constexpr (EPI == EPI_SIGMOID) return 1.f / (1.f + __expf(-x));
  if constexpr (EPI == EPI_RELUSQ)  { float t = fmaxf(x, 0.f); return t * t; }
  if constexpr (EPI == EPI_EEW)     { float w = e0[gn] + x; return __expf(-__expf(w)); }
  if constexpr (EPI == EPI_RESADD)  return e0[(size_t)gm * N + gn] + x;
  if constexpr (EPI == EPI_FINAL)   return e0[(size_t)gm * N + gn] + e1[(size_t)gm * N + gn] * x;
  return x;
}

// ---------------------------------------------------------------------------
// GEMM: out[M,N] = epi( A[M,K] @ B[K,N] ), bf16 WMMA, fp32 accumulate.
// NK==1: B[k][n] = Wt[n*K + k]  (x @ W.T, row-major W[N,K]) -- vectorized path
// NK==0: B[k][n] = Wt[k*N + n]  (row-major W[K,N])          -- scalar path
// Requirements (met by all call sites): M % 128 == 0, K % 32 == 0,
// 16B-aligned pointers; NK==1 call sites also have N % 64 == 0.
// Block: 256 threads = 8 waves; tile 128x64, BK=32; 4 wmma per wave per step.
// ---------------------------------------------------------------------------
#define BM 128
#define BN 64
#define BK 32
#define LDT 40   // ushort row stride: 80 bytes, 16B aligned

template <int NK, int EPI>
__global__ __launch_bounds__(256)
void gemm_wmma(const float* __restrict__ A, const float* __restrict__ Wt,
               float* __restrict__ out, int M, int N, int K,
               const float* __restrict__ e0, const float* __restrict__ e1) {
  __shared__ unsigned short As[BM * LDT];   // [m][k], bf16 bits
  __shared__ unsigned short Bs[BN * LDT];   // [n][k], bf16 bits

  const int tid  = threadIdx.x;
  const int wave = tid >> 5;
  const int lane = tid & 31;
  const int m0 = blockIdx.y * BM;
  const int n0 = blockIdx.x * BN;
  const int wm = (wave & 3) * 32;           // wave macro-tile: 32x32
  const int wn = (wave >> 2) * 32;
  const int half = lane >> 4;
  const int l15  = lane & 15;

  const int arow = tid >> 3, ac4 = (tid & 7) << 2;   // A/B staging coords

  v8f acc00 = {}, acc01 = {}, acc10 = {}, acc11 = {};

  for (int k0 = 0; k0 < K; k0 += BK) {
    // ---- Issue all tile loads first (keeps 6 b128 loads in flight) ----
    float4 av[4];
    #pragma unroll
    for (int j = 0; j < 4; ++j)
      av[j] = *(const float4*)(A + (size_t)(m0 + arow + j * 32) * K + k0 + ac4);
    float4 bv[2];
    float  bs[8];
    if (NK) {
      #pragma unroll
      for (int j = 0; j < 2; ++j)
        bv[j] = *(const float4*)(Wt + (size_t)(n0 + arow + j * 32) * K + k0 + ac4);
    } else {
      #pragma unroll
      for (int j = 0; j < 8; ++j) {          // 2048 scalars, coalesced in n
        int idx = tid + j * 256;
        int gn = n0 + (idx & 63);
        bs[j] = (gn < N) ? Wt[(size_t)(k0 + (idx >> 6)) * N + gn] : 0.f;
      }
    }
    // Prefetch next K-tile (speculative; gfx1250 global_prefetch_b8)
    if (k0 + BK < K) {
      __builtin_prefetch(A + (size_t)(m0 + (tid >> 1)) * K + k0 + BK, 0, 1);
      if (NK) __builtin_prefetch(Wt + (size_t)(n0 + (tid & 63)) * K + k0 + BK, 0, 1);
      else    __builtin_prefetch(Wt + (size_t)(k0 + BK) * N + n0 + (tid & 63), 0, 1);
    }
    // ---- Convert + store to LDS ----
    #pragma unroll
    for (int j = 0; j < 4; ++j)
      st4bf(&As[(arow + j * 32) * LDT + ac4], av[j]);
    if (NK) {
      #pragma unroll
      for (int j = 0; j < 2; ++j)
        st4bf(&Bs[(arow + j * 32) * LDT + ac4], bv[j]);
    } else {
      #pragma unroll
      for (int j = 0; j < 8; ++j) {
        int idx = tid + j * 256;
        Bs[(idx & 63) * LDT + (idx >> 6)] = f2bf(bs[j]);
      }
    }
    __syncthreads();

    // ---- Fragments per ISA layout ----
    // A 16x32: lane row = M%16, K(e) = (e/8)*16 + half*8 + ((e/2)%4)*2 + e%2
    v16bf a0, a1;
    {
      const int ml0 = wm + l15;
      const int ml1 = ml0 + 16;
      #pragma unroll
      for (int e = 0; e < 16; ++e) {
        int kk = ((e >> 3) << 4) + (half << 3) + (((e >> 1) & 3) << 1) + (e & 1);
        a0[e] = bf_bits(As[ml0 * LDT + kk]);
        a1[e] = bf_bits(As[ml1 * LDT + kk]);
      }
    }
    // B 32x16: lane col = N%16, K(e) = (e/8)*16 + half*8 + e%8
    v16bf b0, b1;
    {
      const int nl0 = wn + l15;
      const int nl1 = nl0 + 16;
      #pragma unroll
      for (int e = 0; e < 16; ++e) {
        int kk = ((e >> 3) << 4) + (half << 3) + (e & 7);
        b0[e] = bf_bits(Bs[nl0 * LDT + kk]);
        b1[e] = bf_bits(Bs[nl1 * LDT + kk]);
      }
    }
    acc00 = __builtin_amdgcn_wmma_f32_16x16x32_bf16(false, a0, false, b0, (short)0, acc00, false, false);
    acc01 = __builtin_amdgcn_wmma_f32_16x16x32_bf16(false, a0, false, b1, (short)0, acc01, false, false);
    acc10 = __builtin_amdgcn_wmma_f32_16x16x32_bf16(false, a1, false, b0, (short)0, acc10, false, false);
    acc11 = __builtin_amdgcn_wmma_f32_16x16x32_bf16(false, a1, false, b1, (short)0, acc11, false, false);
    __syncthreads();
  }

  // ---- Store: D VGPR v -> (M = v + 8*half, N = lane%16) ----
  v8f accs[2][2] = {{acc00, acc01}, {acc10, acc11}};
  #pragma unroll
  for (int ti = 0; ti < 2; ++ti) {
    #pragma unroll
    for (int tj = 0; tj < 2; ++tj) {
      #pragma unroll
      for (int v = 0; v < 8; ++v) {
        int gm = m0 + wm + ti * 16 + v + 8 * half;
        int gn = n0 + wn + tj * 16 + l15;
        if (gn < N)
          out[(size_t)gm * N + gn] =
              apply_epi<EPI>(accs[ti][tj][v], gm, gn, N, e0, e1);
      }
    }
  }
}

// ---------------------------------------------------------------------------
// Block-wide sum over 256 threads.
// ---------------------------------------------------------------------------
__device__ __forceinline__ float block_sum(float v, float* red) {
  int tid = threadIdx.x;
  red[tid] = v;
  __syncthreads();
  for (int s = 128; s > 0; s >>= 1) {
    if (tid < s) red[tid] += red[tid + s];
    __syncthreads();
  }
  float r = red[0];
  __syncthreads();
  return r;
}

// LayerNorm over rows of [M, C].
__global__ __launch_bounds__(256)
void ln_rows(const float* __restrict__ x, const float* __restrict__ w,
             const float* __restrict__ b, float* __restrict__ out) {
  __shared__ float red[256];
  const int row = blockIdx.x;
  const int c0 = threadIdx.x * 8;
  const float* xr = x + (size_t)row * kC;
  float loc[8];
  float s = 0.f;
  #pragma unroll
  for (int i = 0; i < 8; ++i) { loc[i] = xr[c0 + i]; s += loc[i]; }
  float mean = block_sum(s, red) * (1.f / kC);
  float sq = 0.f;
  #pragma unroll
  for (int i = 0; i < 8; ++i) { float d = loc[i] - mean; sq += d * d; }
  float rstd = rsqrtf(block_sum(sq, red) * (1.f / kC) + 1e-5f);
  float* o = out + (size_t)row * kC;
  #pragma unroll
  for (int i = 0; i < 8; ++i)
    o[c0 + i] = (loc[i] - mean) * rstd * w[c0 + i] + b[c0 + i];
}

// LN of row t and row t-1 (zero if t==0), then token-shift mixing.
// mode 0 (tmix): out0=xa, out1=xx=prev-xa, out2=xa+xx*coef0
// mode 1 (cmix): out0=xf+xx*coef0, out1=xf+xx*coef1
__global__ __launch_bounds__(256)
void prep_kernel(const float* __restrict__ src, const float* __restrict__ lw,
                 const float* __restrict__ lb, const float* __restrict__ coef0,
                 const float* __restrict__ coef1,
                 float* __restrict__ out0, float* __restrict__ out1,
                 float* __restrict__ out2, int mode) {
  __shared__ float red[256];
  const int row = blockIdx.x;
  const int t = row % kT;                      // block-uniform
  const int c0 = threadIdx.x * 8;
  const float* xr = src + (size_t)row * kC;

  float cur[8], prv[8];
  float s = 0.f;
  #pragma unroll
  for (int i = 0; i < 8; ++i) { cur[i] = xr[c0 + i]; s += cur[i]; }
  float mean = block_sum(s, red) * (1.f / kC);
  float sq = 0.f;
  #pragma unroll
  for (int i = 0; i < 8; ++i) { float d = cur[i] - mean; sq += d * d; }
  float rstd = rsqrtf(block_sum(sq, red) * (1.f / kC) + 1e-5f);
  #pragma unroll
  for (int i = 0; i < 8; ++i)
    cur[i] = (cur[i] - mean) * rstd * lw[c0 + i] + lb[c0 + i];

  if (t > 0) {                                  // uniform branch: barriers safe
    const float* xp = xr - kC;
    float sp = 0.f;
    #pragma unroll
    for (int i = 0; i < 8; ++i) { prv[i] = xp[c0 + i]; sp += prv[i]; }
    float mp = block_sum(sp, red) * (1.f / kC);
    float qp = 0.f;
    #pragma unroll
    for (int i = 0; i < 8; ++i) { float d = prv[i] - mp; qp += d * d; }
    float rp = rsqrtf(block_sum(qp, red) * (1.f / kC) + 1e-5f);
    #pragma unroll
    for (int i = 0; i < 8; ++i)
      prv[i] = (prv[i] - mp) * rp * lw[c0 + i] + lb[c0 + i];
  } else {
    #pragma unroll
    for (int i = 0; i < 8; ++i) prv[i] = 0.f;
  }

  const size_t base = (size_t)row * kC + c0;
  #pragma unroll
  for (int i = 0; i < 8; ++i) {
    float xav = cur[i];
    float xxv = prv[i] - xav;
    if (mode == 0) {
      out0[base + i] = xav;
      out1[base + i] = xxv;
      out2[base + i] = xav + xxv * coef0[c0 + i];
    } else {
      out0[base + i] = xav + xxv * coef0[c0 + i];
      out1[base + i] = xav + xxv * coef1[c0 + i];
    }
  }
}

// mix_f = xa + xx * (maa_wkvrg[f] + t5[row,f,:] @ tm_w2[f])   for f=0..4
__global__ __launch_bounds__(256)
void mix_kernel(const float* __restrict__ t5, const float* __restrict__ tm_w2,
                const float* __restrict__ maa, const float* __restrict__ xa,
                const float* __restrict__ xx, float* __restrict__ mixb) {
  __shared__ float t5s[160];
  const int row = blockIdx.x;
  if (threadIdx.x < 160) t5s[threadIdx.x] = t5[(size_t)row * 160 + threadIdx.x];
  __syncthreads();
  for (int j = 0; j < 8; ++j) {
    int c = threadIdx.x + j * 256;
    size_t idx = (size_t)row * kC + c;
    float xav = xa[idx], xxv = xx[idx];
    #pragma unroll
    for (int f = 0; f < 5; ++f) {
      float acc = maa[f * kC + c];
      const float* w2 = tm_w2 + ((size_t)f * 32) * kC + c;
      #pragma unroll
      for (int e = 0; e < 32; ++e) acc += t5s[f * 32 + e] * w2[(size_t)e * kC];
      mixb[(size_t)f * kMC + idx] = xav + xxv * acc;
    }
  }
}

// WKV scan: one block per (b,h); 64 threads; thread m owns state column S[:,m].
__global__ __launch_bounds__(64)
void wkv_scan(const float* __restrict__ r, const float* __restrict__ k,
              const float* __restrict__ v, const float* __restrict__ w,
              const float* __restrict__ u, float* __restrict__ y) {
  const int bh = blockIdx.x;
  const int b = bh / kH, h = bh % kH;
  const int m = threadIdx.x;

  __shared__ float rs[64], ks[64], ws[64], vs[64], us[64];
  us[m] = u[h * kHS + m];

  float S[64];
  #pragma unroll
  for (int n = 0; n < 64; ++n) S[n] = 0.f;

  const size_t base = (size_t)b * kT * kC + (size_t)h * kHS;
  const float* rb = r + base;
  const float* kb = k + base;
  const float* vb = v + base;
  const float* wb = w + base;
  float* yb = y + base;

  for (int t = 0; t < kT; ++t) {
    __syncthreads();
    rs[m] = rb[m]; ks[m] = kb[m]; ws[m] = wb[m]; vs[m] = vb[m];
    __syncthreads();
    float vm = vs[m];
    float ruk = 0.f;
    #pragma unroll
    for (int n = 0; n < 64; ++n) ruk += rs[n] * us[n] * ks[n];
    float acc = ruk * vm;
    #pragma unroll
    for (int n = 0; n < 64; ++n) {
      acc += rs[n] * S[n];
      S[n] = S[n] * ws[n] + ks[n] * vm;
    }
    yb[m] = acc;
    rb += kC; kb += kC; vb += kC; wb += kC; yb += kC;
  }
}

// GroupNorm (n_head groups of 64) then multiply by gate g.
__global__ __launch_bounds__(256)
void gn_kernel(const float* __restrict__ y, const float* __restrict__ g,
               const float* __restrict__ lw, const float* __restrict__ lb,
               float* __restrict__ ygn) {
  __shared__ float hsum[kH], hsq[kH];
  const int row = blockIdx.x;
  if (threadIdx.x < kH) { hsum[threadIdx.x] = 0.f; hsq[threadIdx.x] = 0.f; }
  __syncthreads();
  const int c0 = threadIdx.x * 8;
  const int h = c0 >> 6;
  float vals[8];
  float s = 0.f, sq = 0.f;
  #pragma unroll
  for (int i = 0; i < 8; ++i) {
    vals[i] = y[(size_t)row * kC + c0 + i];
    s += vals[i]; sq += vals[i] * vals[i];
  }
  atomicAdd(&hsum[h], s);
  atomicAdd(&hsq[h], sq);
  __syncthreads();
  float mean = hsum[h] * (1.f / kHS);
  float var = hsq[h] * (1.f / kHS) - mean * mean;
  float rstd = rsqrtf(var + 6.4e-4f);          // 1e-5 * 8^2
  #pragma unroll
  for (int i = 0; i < 8; ++i) {
    float yn = (vals[i] - mean) * rstd * lw[c0 + i] + lb[c0 + i];
    ygn[(size_t)row * kC + c0 + i] = yn * g[(size_t)row * kC + c0 + i];
  }
}

// ---------------------------------------------------------------------------
extern "C" void kernel_launch(void* const* d_in, const int* in_sizes, int n_in,
                              void* d_out, int out_size, void* d_ws, size_t ws_size,
                              hipStream_t stream) {
  (void)in_sizes; (void)n_in; (void)out_size; (void)ws_size;

  const float* x        = (const float*)d_in[0];
  const float* ln0_w    = (const float*)d_in[1];
  const float* ln0_b    = (const float*)d_in[2];
  const float* ln1_w    = (const float*)d_in[3];
  const float* ln1_b    = (const float*)d_in[4];
  const float* ln2_w    = (const float*)d_in[5];
  const float* ln2_b    = (const float*)d_in[6];
  const float* maa_x    = (const float*)d_in[7];
  const float* maa_wkvrg= (const float*)d_in[8];
  const float* tm_w1    = (const float*)d_in[9];
  const float* tm_w2    = (const float*)d_in[10];
  const float* td       = (const float*)d_in[11];
  const float* td_w1    = (const float*)d_in[12];
  const float* td_w2    = (const float*)d_in[13];
  const float* u        = (const float*)d_in[14];
  const float* Wr       = (const float*)d_in[15];
  const float* Wk       = (const float*)d_in[16];
  const float* Wv       = (const float*)d_in[17];
  const float* Wg       = (const float*)d_in[18];
  const float* Wo       = (const float*)d_in[19];
  const float* lnx_w    = (const float*)d_in[20];
  const float* lnx_b    = (const float*)d_in[21];
  const float* cmaa_k   = (const float*)d_in[22];
  const float* cmaa_r   = (const float*)d_in[23];
  const float* Wfk      = (const float*)d_in[24];
  const float* Wfr      = (const float*)d_in[25];
  const float* Wfv      = (const float*)d_in[26];

  float* ws = (float*)d_ws;
  auto slot = [&](int i) { return ws + (size_t)i * kMC; };
  float* xbuf = slot(0);                 // ln0(x), residual for Wo
  float* xa   = slot(1);
  float* xxb  = slot(2);
  float* xxx  = slot(3);
  float* mixb = slot(4);                 // 5 slots: xw,xk,xv,xr,xg
  float* rb   = slot(9);
  float* kb   = slot(10);
  float* vb   = slot(11);
  float* gb   = slot(12);
  float* eew  = slot(13);
  float* yb   = slot(14);
  float* ygn  = slot(15);
  float* t5   = ws + 16 * kMC;           // [8192,160]
  float* t64  = t5 + (size_t)kM * 160;   // [8192,64]
  // Reuse after Tmix:
  float* x2   = slot(3);                 // residual stream after Tmix
  float* kf   = slot(4);                 // [8192,7168] spans slots 4..7.5
  float* xk2  = slot(9);
  float* xr2  = slot(10);
  float* rr   = slot(11);

  auto gemm = [&](const float* A, const float* Wt, float* out,
                  int M, int N, int K, int nkf, int epi,
                  const float* e0, const float* e1) {
    dim3 grid((N + BN - 1) / BN, M / BM);
    #define GCASE(NKv, EPIv) \
      gemm_wmma<NKv, EPIv><<<grid, 256, 0, stream>>>(A, Wt, out, M, N, K, e0, e1)
    if (nkf) {
      switch (epi) {
        case EPI_NONE:    GCASE(1, EPI_NONE);    break;
        case EPI_SILU:    GCASE(1, EPI_SILU);    break;
        case EPI_SIGMOID: GCASE(1, EPI_SIGMOID); break;
        case EPI_RELUSQ:  GCASE(1, EPI_RELUSQ);  break;
        case EPI_RESADD:  GCASE(1, EPI_RESADD);  break;
        case EPI_FINAL:   GCASE(1, EPI_FINAL);   break;
        default:          GCASE(1, EPI_NONE);    break;
      }
    } else {
      switch (epi) {
        case EPI_TANH:    GCASE(0, EPI_TANH);    break;
        case EPI_EEW:     GCASE(0, EPI_EEW);     break;
        default:          GCASE(0, EPI_NONE);    break;
      }
    }
    #undef GCASE
  };

  // ---- Tmix ----
  ln_rows<<<kM, 256, 0, stream>>>(x, ln0_w, ln0_b, xbuf);
  prep_kernel<<<kM, 256, 0, stream>>>(xbuf, ln1_w, ln1_b, maa_x, nullptr,
                                      xa, xxb, xxx, 0);
  gemm(xxx, tm_w1, t5, kM, 160, kC, 0, EPI_TANH, nullptr, nullptr);
  mix_kernel<<<kM, 256, 0, stream>>>(t5, tm_w2, maa_wkvrg, xa, xxb, mixb);

  gemm(mixb + 3 * kMC, Wr, rb, kM, kC, kC, 1, EPI_NONE, nullptr, nullptr); // r
  gemm(mixb + 1 * kMC, Wk, kb, kM, kC, kC, 1, EPI_NONE, nullptr, nullptr); // k
  gemm(mixb + 2 * kMC, Wv, vb, kM, kC, kC, 1, EPI_NONE, nullptr, nullptr); // v
  gemm(mixb + 4 * kMC, Wg, gb, kM, kC, kC, 1, EPI_SILU, nullptr, nullptr); // g
  gemm(mixb + 0 * kMC, td_w1, t64, kM, 64, kC, 0, EPI_TANH, nullptr, nullptr);
  gemm(t64, td_w2, eew, kM, kC, 64, 0, EPI_EEW, td, nullptr);              // exp(-exp(w))

  wkv_scan<<<kB * kH, 64, 0, stream>>>(rb, kb, vb, eew, u, yb);
  gn_kernel<<<kM, 256, 0, stream>>>(yb, gb, lnx_w, lnx_b, ygn);
  gemm(ygn, Wo, x2, kM, kC, kC, 1, EPI_RESADD, xbuf, nullptr);             // x += .@Wo.T

  // ---- CMix ----
  prep_kernel<<<kM, 256, 0, stream>>>(x2, ln2_w, ln2_b, cmaa_k, cmaa_r,
                                      xk2, xr2, nullptr, 1);
  gemm(xk2, Wfk, kf, kM, kFF, kC, 1, EPI_RELUSQ, nullptr, nullptr);
  gemm(xr2, Wfr, rr, kM, kC, kC, 1, EPI_SIGMOID, nullptr, nullptr);
  gemm(kf, Wfv, (float*)d_out, kM, kC, kFF, 1, EPI_FINAL, x2, rr);
}
